// MessagePassingLayer_15006615732841
// MI455X (gfx1250) — matrix operand order repaired and checked by reading, not compiled, not run on hardware
//
#include <hip/hip_runtime.h>

#define NODE_DIM 128
#define EDGE_DIM 64
#define OUT_DIM  128
#define N_NODES  50000
#define N_EDGES  800000

typedef __bf16 bf16_t;
typedef bf16_t v16bf __attribute__((ext_vector_type(16)));
typedef bf16_t v8bf  __attribute__((ext_vector_type(8)));
typedef float  v8f   __attribute__((ext_vector_type(8)));

// ---------------------------------------------------------------------------
// WMMA wrapper: D = A(16x32 bf16) * B(32x16 bf16) + C(16x16 f32)
// ---------------------------------------------------------------------------
static __device__ inline v8f wmma_bf16(v16bf a, v16bf b, v8f c) {
  return __builtin_amdgcn_wmma_f32_16x16x32_bf16(
      /*neg_a=*/false, a, /*neg_b=*/false, b,
      /*c_mod=*/(short)0, c, /*reuse_a=*/false, /*reuse_b=*/false);
}

// Build a 16-element bf16 A fragment from f32 memory.
// Per ISA 7.12.2 (16-bit A 16x32): lanes 0-15 hold K = c..c+7 and c+16..c+23,
// lanes 16-31 hold K = c+8..c+15 and c+24..c+31 -> caller passes c0 already
// shifted by +8 for hi lanes; elements 0..7 at c0, 8..15 at c0+16.
static __device__ inline v16bf frag_f32(const float* base, int c0) {
  float4 f0 = *(const float4*)(base + c0);
  float4 f1 = *(const float4*)(base + c0 + 4);
  float4 f2 = *(const float4*)(base + c0 + 16);
  float4 f3 = *(const float4*)(base + c0 + 20);
  v16bf a;
  a[0]=(bf16_t)f0.x;  a[1]=(bf16_t)f0.y;  a[2]=(bf16_t)f0.z;  a[3]=(bf16_t)f0.w;
  a[4]=(bf16_t)f1.x;  a[5]=(bf16_t)f1.y;  a[6]=(bf16_t)f1.z;  a[7]=(bf16_t)f1.w;
  a[8]=(bf16_t)f2.x;  a[9]=(bf16_t)f2.y;  a[10]=(bf16_t)f2.z; a[11]=(bf16_t)f2.w;
  a[12]=(bf16_t)f3.x; a[13]=(bf16_t)f3.y; a[14]=(bf16_t)f3.z; a[15]=(bf16_t)f3.w;
  return a;
}

static __device__ inline v16bf frag_f32s(const float* base, int c0, float s) {
  float4 f0 = *(const float4*)(base + c0);
  float4 f1 = *(const float4*)(base + c0 + 4);
  float4 f2 = *(const float4*)(base + c0 + 16);
  float4 f3 = *(const float4*)(base + c0 + 20);
  v16bf a;
  a[0]=(bf16_t)(f0.x*s);  a[1]=(bf16_t)(f0.y*s);  a[2]=(bf16_t)(f0.z*s);  a[3]=(bf16_t)(f0.w*s);
  a[4]=(bf16_t)(f1.x*s);  a[5]=(bf16_t)(f1.y*s);  a[6]=(bf16_t)(f1.z*s);  a[7]=(bf16_t)(f1.w*s);
  a[8]=(bf16_t)(f2.x*s);  a[9]=(bf16_t)(f2.y*s);  a[10]=(bf16_t)(f2.z*s); a[11]=(bf16_t)(f2.w*s);
  a[12]=(bf16_t)(f3.x*s); a[13]=(bf16_t)(f3.y*s); a[14]=(bf16_t)(f3.z*s); a[15]=(bf16_t)(f3.w*s);
  return a;
}

// Native CDNA5 fp atomics (no return -> STOREcnt), bypassing CAS fallbacks.
static __device__ inline void gatomic_add_f32(float* p, float v) {
  asm volatile("global_atomic_add_f32 %0, %1, off" : : "v"(p), "v"(v) : "memory");
}
static __device__ inline void gatomic_max_f32(float* p, float v) {
  asm volatile("global_atomic_max_num_f32 %0, %1, off" : : "v"(p), "v"(v) : "memory");
}

// ---------------------------------------------------------------------------
// Kernel 1: init aggregation buffers (must run every launch; determinism)
// ---------------------------------------------------------------------------
__global__ __launch_bounds__(256) void init_kernel(float* __restrict__ sumbuf,
                                                   float* __restrict__ maxbuf,
                                                   unsigned* __restrict__ cnt) {
  long long i = (long long)blockIdx.x * 256 + threadIdx.x;
  const long long NN = (long long)N_NODES * OUT_DIM;
  if (i < NN) { sumbuf[i] = 0.0f; maxbuf[i] = -3.402823466e38f; }
  if (i < N_NODES) cnt[i] = 0u;
}

// ---------------------------------------------------------------------------
// Kernel 2: pack/transpose weights to bf16 (column-major per output col so a
// lane's K-values form two contiguous runs of 8 -> two b128 loads per B fragment)
// ---------------------------------------------------------------------------
__global__ __launch_bounds__(256) void pack_kernel(
    const float* __restrict__ W1, const float* __restrict__ W2,
    const float* __restrict__ Wm, const float* __restrict__ Wx,
    const float* __restrict__ Ws, const float* __restrict__ bm,
    const float* __restrict__ bx, const float* __restrict__ bs,
    bf16_t* __restrict__ W1t, bf16_t* __restrict__ W2t,
    bf16_t* __restrict__ Wct, float* __restrict__ btot) {
  int i = blockIdx.x * 256 + threadIdx.x;
  if (i < 128 * 192) { int n = i / 192, k = i % 192; W1t[i] = (bf16_t)W1[k * 128 + n]; }
  if (i < 128 * 128) { int n = i / 128, k = i % 128; W2t[i] = (bf16_t)W2[k * 128 + n]; }
  if (i < 128 * 384) {
    int n = i / 384, k = i % 384;
    float v = (k < 128) ? Wm[k * 128 + n]
            : (k < 256) ? Wx[(k - 128) * 128 + n]
                        : Ws[(k - 256) * 128 + n];
    Wct[i] = (bf16_t)v;
  }
  if (i < 128) btot[i] = bm[i] + bx[i] + bs[i];
}

// ---------------------------------------------------------------------------
// Kernel 3: fused edge MLP (two bf16 WMMA GEMMs) + scatter aggregation
// One wave = 16 edges. 8 waves / block. 800000/16 = 50000 tiles exactly.
// ---------------------------------------------------------------------------
__global__ __launch_bounds__(256) void edge_kernel(
    const float* __restrict__ x, const long long* __restrict__ ei,
    const float* __restrict__ ea, const float* __restrict__ b1,
    const float* __restrict__ b2, const bf16_t* __restrict__ W1t,
    const bf16_t* __restrict__ W2t, float* __restrict__ sumbuf,
    float* __restrict__ maxbuf, unsigned* __restrict__ cnt) {
  __shared__ bf16_t hbuf[8][16][128];   // per-wave C->A relayout staging (32KB)
  __shared__ int dstbuf[8][16];

  const int wave = threadIdx.x >> 5;
  const int lane = threadIdx.x & 31;
  const int lrow = lane & 15;
  const int khi  = lane >> 4;           // 0 for lanes 0-15, 1 for 16-31

  const int tile = blockIdx.x * 8 + wave;
  const int e0   = tile * 16;

  const long long* dstp = ei + N_EDGES; // row 1 of edge_index = dst
  const int myDst = (int)dstp[e0 + lrow];
  if (!khi) {
    dstbuf[wave][lrow] = myDst;
    atomicAdd(&cnt[myDst], 1u);         // global_atomic_add_u32
  }

  // Prefetch upcoming edge_attr stream (CDNA5 global_prefetch_b8).
  if (e0 + 2048 < N_EDGES)
    __builtin_prefetch(ea + (long long)(e0 + 2048) * EDGE_DIM, 0, 1);

  // ---- GEMM1 A fragments: combined = [x[dst] (K 0..127) | edge_attr (K 128..191)]
  const float* xrow = x + (long long)myDst * NODE_DIM;
  const float* erow = ea + (long long)(e0 + lrow) * EDGE_DIM;
  v16bf a1[6];
#pragma unroll
  for (int k = 0; k < 4; ++k) a1[k] = frag_f32(xrow, k * 32 + khi * 8);
#pragma unroll
  for (int k = 0; k < 2; ++k) a1[4 + k] = frag_f32(erow, k * 32 + khi * 8);

  // ---- GEMM1: h = relu(combined @ W1 + b1), 8 N-tiles x 6 K-steps = 48 WMMAs
#pragma unroll
  for (int nt = 0; nt < 8; ++nt) {
    const int n = nt * 16 + lrow;
    const bf16_t* wrow = W1t + n * 192;
    v8f acc = {0, 0, 0, 0, 0, 0, 0, 0};
#pragma unroll
    for (int k = 0; k < 6; ++k) {
      v8bf blo = *(const v8bf*)(wrow + k * 32 + khi * 8);
      v8bf bhi = *(const v8bf*)(wrow + k * 32 + khi * 8 + 16);
      v16bf b = __builtin_shufflevector(blo, bhi, 0, 1, 2, 3, 4, 5, 6, 7, 8, 9, 10, 11, 12, 13, 14, 15);
      acc = wmma_bf16(a1[k], b, acc);
    }
    const float bias = b1[n];
#pragma unroll
    for (int r = 0; r < 8; ++r) {
      float h = acc[r] + bias;
      h = h > 0.0f ? h : 0.0f;
      hbuf[wave][r + khi * 8][n] = (bf16_t)h;   // C layout -> row-major LDS
    }
  }

  // ---- Reload h in A layout (same wave; DS ordering handles the hazard)
  v16bf a2[4];
#pragma unroll
  for (int k = 0; k < 4; ++k) {
    const int c0 = k * 32 + khi * 8;
    v8bf lo = *(const v8bf*)&hbuf[wave][lrow][c0];
    v8bf hi = *(const v8bf*)&hbuf[wave][lrow][c0 + 16];
    a2[k] = __builtin_shufflevector(lo, hi, 0, 1, 2, 3, 4, 5, 6, 7,
                                    8, 9, 10, 11, 12, 13, 14, 15);
  }

  int nodes[8];
#pragma unroll
  for (int r = 0; r < 8; ++r) nodes[r] = dstbuf[wave][r + khi * 8];

  // ---- GEMM2: messages = h @ W2 + b2, then scatter sum/max atomics
#pragma unroll
  for (int nt = 0; nt < 8; ++nt) {
    const int n = nt * 16 + lrow;
    const bf16_t* wrow = W2t + n * 128;
    v8f acc = {0, 0, 0, 0, 0, 0, 0, 0};
#pragma unroll
    for (int k = 0; k < 4; ++k) {
      v8bf blo = *(const v8bf*)(wrow + k * 32 + khi * 8);
      v8bf bhi = *(const v8bf*)(wrow + k * 32 + khi * 8 + 16);
      v16bf b = __builtin_shufflevector(blo, bhi, 0, 1, 2, 3, 4, 5, 6, 7, 8, 9, 10, 11, 12, 13, 14, 15);
      acc = wmma_bf16(a2[k], b, acc);
    }
    const float bias = b2[n];
#pragma unroll
    for (int r = 0; r < 8; ++r) {
      const float m = acc[r] + bias;
      float* base = sumbuf + (long long)nodes[r] * OUT_DIM + n;
      float* baseM = maxbuf + (long long)nodes[r] * OUT_DIM + n;
      gatomic_add_f32(base, m);
      gatomic_max_f32(baseM, m);
    }
  }
}

// ---------------------------------------------------------------------------
// Kernel 4: out = mean@Wm + max@Wx + sum@Ws + (bm+bx+bs), fused as
// [mean|max|sum] (16x384) @ Wcat (384x128). One wave = 16 nodes.
// ---------------------------------------------------------------------------
__global__ __launch_bounds__(256) void node_kernel(
    const float* __restrict__ sumbuf, const float* __restrict__ maxbuf,
    const unsigned* __restrict__ cnt, const bf16_t* __restrict__ Wct,
    const float* __restrict__ btot, float* __restrict__ out) {
  const int wave = threadIdx.x >> 5;
  const int lane = threadIdx.x & 31;
  const int lrow = lane & 15;
  const int khi  = lane >> 4;

  const int tile = blockIdx.x * 8 + wave;
  if (tile >= N_NODES / 16) return;     // wave-uniform exit; EXEC stays full
  const int n0 = tile * 16;

  const int node = n0 + lrow;
  const unsigned c = cnt[node];
  const float inv = 1.0f / (float)(c > 0u ? c : 1u);  // mean divisor
  const float msc = c > 0u ? 1.0f : 0.0f;             // empty-segment max -> 0

  const float* srow = sumbuf + (long long)node * OUT_DIM;
  const float* mrow = maxbuf + (long long)node * OUT_DIM;

  v16bf a[12];
#pragma unroll
  for (int k = 0; k < 4; ++k) a[k]     = frag_f32s(srow, k * 32 + khi * 8, inv);
#pragma unroll
  for (int k = 0; k < 4; ++k) a[4 + k] = frag_f32s(mrow, k * 32 + khi * 8, msc);
#pragma unroll
  for (int k = 0; k < 4; ++k) a[8 + k] = frag_f32(srow, k * 32 + khi * 8);

#pragma unroll
  for (int nt = 0; nt < 8; ++nt) {
    const int n = nt * 16 + lrow;
    const bf16_t* wrow = Wct + n * 384;
    v8f acc = {0, 0, 0, 0, 0, 0, 0, 0};
#pragma unroll
    for (int k = 0; k < 12; ++k) {
      v8bf blo = *(const v8bf*)(wrow + k * 32 + khi * 8);
      v8bf bhi = *(const v8bf*)(wrow + k * 32 + khi * 8 + 16);
      v16bf b = __builtin_shufflevector(blo, bhi, 0, 1, 2, 3, 4, 5, 6, 7, 8, 9, 10, 11, 12, 13, 14, 15);
      acc = wmma_bf16(a[k], b, acc);
    }
    const float bias = btot[n];
#pragma unroll
    for (int r = 0; r < 8; ++r) {
      const int nr = n0 + r + khi * 8;
      out[(long long)nr * OUT_DIM + n] = acc[r] + bias;
    }
  }
}

// ---------------------------------------------------------------------------
// Host launcher
// ---------------------------------------------------------------------------
extern "C" void kernel_launch(void* const* d_in, const int* in_sizes, int n_in,
                              void* d_out, int out_size, void* d_ws, size_t ws_size,
                              hipStream_t stream) {
  const float*     x  = (const float*)d_in[0];
  const long long* ei = (const long long*)d_in[1];
  const float*     ea = (const float*)d_in[2];
  const float*     W1 = (const float*)d_in[3];
  const float*     b1 = (const float*)d_in[4];
  const float*     W2 = (const float*)d_in[5];
  const float*     b2 = (const float*)d_in[6];
  const float*     Wm = (const float*)d_in[7];
  const float*     bm = (const float*)d_in[8];
  const float*     Wx = (const float*)d_in[9];
  const float*     bx = (const float*)d_in[10];
  const float*     Ws = (const float*)d_in[11];
  const float*     bs = (const float*)d_in[12];
  float* out = (float*)d_out;

  char* ws = (char*)d_ws;
  size_t off = 0;
  auto take = [&](size_t bytes) -> char* {
    char* p = ws + off;
    off = (off + bytes + 255) & ~(size_t)255;
    return p;
  };
  float*    sumbuf = (float*)take((size_t)N_NODES * OUT_DIM * 4);
  float*    maxbuf = (float*)take((size_t)N_NODES * OUT_DIM * 4);
  unsigned* cnt    = (unsigned*)take((size_t)N_NODES * 4);
  bf16_t*   W1t    = (bf16_t*)take((size_t)128 * 192 * 2);
  bf16_t*   W2t    = (bf16_t*)take((size_t)128 * 128 * 2);
  bf16_t*   Wct    = (bf16_t*)take((size_t)128 * 384 * 2);
  float*    btot   = (float*)take(128 * 4);

  // 1) init aggregation state (every call — harness doesn't re-zero d_ws)
  const long long NN = (long long)N_NODES * OUT_DIM;
  init_kernel<<<(int)((NN + 255) / 256), 256, 0, stream>>>(sumbuf, maxbuf, cnt);

  // 2) pack weights to transposed bf16
  pack_kernel<<<(128 * 384 + 255) / 256, 256, 0, stream>>>(
      W1, W2, Wm, Wx, Ws, bm, bx, bs, W1t, W2t, Wct, btot);

  // 3) fused edge MLP + scatter (50000 edge tiles, 8 waves/block)
  edge_kernel<<<N_EDGES / 16 / 8, 256, 0, stream>>>(
      x, ei, ea, b1, b2, W1t, W2t, sumbuf, maxbuf, cnt);

  // 4) fused node aggregator linears (3125 node tiles)
  node_kernel<<<(N_NODES / 16 + 7) / 8, 256, 0, stream>>>(
      sumbuf, maxbuf, cnt, Wct, btot, out);
}
